// MultiHeadAttentionBlock_17849884082791
// MI455X (gfx1250) — matrix operand order repaired and checked by reading, not compiled, not run on hardware
//
#include <hip/hip_runtime.h>
#include <hip/hip_bf16.h>

typedef __attribute__((ext_vector_type(16))) _Float16 v16h;
typedef __attribute__((ext_vector_type(8)))  _Float16 v8h;
typedef __attribute__((ext_vector_type(8)))  float    v8f;
typedef __attribute__((ext_vector_type(4)))  float    v4f;
typedef __attribute__((ext_vector_type(4)))  unsigned int v4u;
typedef __attribute__((ext_vector_type(8)))  int      v8i;
typedef __attribute__((ext_vector_type(4)))  int      v4i;

#define D_MODEL 1024
#define NHEAD   16
#define DK      64
#define BATCH   2
#define SEQ     2048
#define MTOT    (BATCH * SEQ)   // 4096 rows

// ---------- helpers ----------

__device__ __forceinline__ v8f wmma_f16(v16h a, v16h b, v8f c) {
    return __builtin_amdgcn_wmma_f32_16x16x32_f16(false, a, false, b, (short)0, c,
                                                  false, false);
}

// Load an A fragment (16x32, f16) from a row-major [rows][ldk] tile.
// `row` points at the start of this lane's 32-element K chunk.
__device__ __forceinline__ v16h load_afrag(const _Float16* row, int half) {
    union { v16h v; v8h h[2]; } u;
    u.h[0] = *(const v8h*)(row + 16 * half);       // K 0..7   (or 16..23)
    u.h[1] = *(const v8h*)(row + 16 * half + 8);   // K 8..15  (or 24..31)
    return u.v;
}

// Convert 16 contiguous f32 -> 16 contiguous f16 (LDS staging).
__device__ __forceinline__ void cvt_store16(const float* src, _Float16* dst) {
    v4f a = *(const v4f*)(src + 0);
    v4f b = *(const v4f*)(src + 4);
    v4f c = *(const v4f*)(src + 8);
    v4f d = *(const v4f*)(src + 12);
    v8h lo, hi;
#pragma unroll
    for (int i = 0; i < 4; ++i) {
        lo[i]     = (_Float16)a[i];
        lo[i + 4] = (_Float16)b[i];
        hi[i]     = (_Float16)c[i];
        hi[i + 4] = (_Float16)d[i];
    }
    *(v8h*)(dst + 0) = lo;
    *(v8h*)(dst + 8) = hi;
}

// ---------- Tensor Data Mover: 2D tile load (CDNA5 ISA) ----------
// Loads a tile_y x tile_x tile of `dsize_code` elements (0=1B,1=2B,2=4B) from a
// row-major tensor with row stride `stride_elems` into LDS at byte offset
// `lds_addr` (contiguous tile_x-element rows). Wave-level op; EXEC ignored.
__device__ __forceinline__ void tdm_load_2d(unsigned long long gaddr, unsigned lds_addr,
                                            unsigned dsize_code, unsigned tile_x,
                                            unsigned tile_y,
                                            unsigned long long stride_elems) {
    v4u g0;
    g0[0] = 1u;                                                   // count=1, user mode
    g0[1] = lds_addr;                                             // D#.lds_addr
    g0[2] = (unsigned)gaddr;                                      // global_addr[31:0]
    g0[3] = (unsigned)((gaddr >> 32) & 0x01ffffffull) | (2u << 30);  // addr[56:32] | type=2
    v8i g1;
    g1[0] = (int)(dsize_code << 16);            // workgroup_mask=0, data_size
    g1[1] = (int)(tile_x << 16);                // tensor_dim0[15:0] = tile_x
    g1[2] = (int)(tile_y << 16);                // tensor_dim0 hi=0, tensor_dim1[15:0]
    g1[3] = (int)(tile_x << 16);                // tensor_dim1 hi=0, tile_dim0
    g1[4] = (int)tile_y;                        // tile_dim1, tile_dim2=0
    g1[5] = (int)(unsigned)stride_elems;        // tensor_dim0_stride[31:0]
    g1[6] = (int)((stride_elems >> 32) & 0xffffull);  // stride[47:32], dim1_stride lo=0
    g1[7] = 0;
    v4i zero4 = {0, 0, 0, 0};
    v8i zero8 = {0, 0, 0, 0, 0, 0, 0, 0};
    __builtin_amdgcn_tensor_load_to_lds(g0, g1, zero4, zero4, zero8, 0);
}

// ---------- GEMM: Y = X @ W^T + b ----------
// X: [4096 x 1024], W: [1024 x 1024] row-major (N x K), b: [1024]
// FINAL==0: X is f32, Y is f16.   FINAL==1: X is f16 (TDM-staged), Y is f32.
template <int FINAL>
__global__ void __launch_bounds__(256)
gemm_xwT(const void* __restrict__ Xv, const float* __restrict__ W,
         const float* __restrict__ bias, void* __restrict__ Yv) {
    constexpr int Kd = D_MODEL;   // 1024
    constexpr int N  = D_MODEL;   // 1024
    constexpr int BM = 128, BN = 128, KT = 64;

    __shared__ __align__(128) unsigned char SMEM[2 * 128 * KT * 2];  // 32 KB
    _Float16* Alds = (_Float16*)(SMEM);                 // [128][64]
    _Float16* Blds = (_Float16*)(SMEM + 128 * KT * 2);  // [128][64]

    const int nblk  = blockIdx.x % (N / BN);
    const int mblk  = blockIdx.x / (N / BN);
    const int mbase = mblk * BM;
    const int nbase = nblk * BN;

    const int wave = threadIdx.x >> 5;
    const int lane = threadIdx.x & 31;
    const int half = lane >> 4;
    const int ln   = lane & 15;

    // accumulators: 8 column sub-tiles of 16, seeded with bias
    v8f acc[8];
#pragma unroll
    for (int j = 0; j < 8; ++j) {
        float bv = bias[nbase + j * 16 + ln];
#pragma unroll
        for (int r = 0; r < 8; ++r) acc[j][r] = bv;
    }

    const int trow = threadIdx.x >> 1;          // 0..127
    const int tk   = (threadIdx.x & 1) * 32;    // 0 or 32

#pragma unroll 1
    for (int k0 = 0; k0 < Kd; k0 += KT) {
        // ---- stage A tile ----
        if (FINAL) {
            if (threadIdx.x < 32) {
                const _Float16* X = (const _Float16*)Xv;
                tdm_load_2d((unsigned long long)(X + (size_t)mbase * Kd + k0),
                            0u, 1u, KT, BM, Kd);
            }
        } else {
            const float* X = (const float*)Xv;
            const float* asrc = X + (size_t)(mbase + trow) * Kd + k0 + tk;
            cvt_store16(asrc + 0,  &Alds[trow * KT + tk + 0]);
            cvt_store16(asrc + 16, &Alds[trow * KT + tk + 16]);
        }
        // ---- stage B tile (weights, f32 -> f16) ----
        const float* bsrc = W + (size_t)(nbase + trow) * Kd + k0 + tk;
        cvt_store16(bsrc + 0,  &Blds[trow * KT + tk + 0]);
        cvt_store16(bsrc + 16, &Blds[trow * KT + tk + 16]);
        if (k0 + KT < Kd)
            __builtin_prefetch(W + (size_t)(nbase + trow) * Kd + k0 + KT + tk, 0, 1);
        if (FINAL && threadIdx.x < 32) __builtin_amdgcn_s_wait_tensorcnt(0);
        __syncthreads();

        const v16h a0 = load_afrag(&Alds[(wave * 16 + ln) * KT], half);
        const v16h a1 = load_afrag(&Alds[(wave * 16 + ln) * KT + 32], half);
#pragma unroll
        for (int j = 0; j < 8; ++j) {
            const v16h b0 = *(const v16h*)&Blds[(j * 16 + ln) * KT + 16 * half];
            const v16h b1 = *(const v16h*)&Blds[(j * 16 + ln) * KT + 32 + 16 * half];
            acc[j] = wmma_f16(a0, b0, acc[j]);
            acc[j] = wmma_f16(a1, b1, acc[j]);
        }
        __syncthreads();
    }

#pragma unroll
    for (int j = 0; j < 8; ++j) {
#pragma unroll
        for (int r = 0; r < 8; ++r) {
            const int row = mbase + wave * 16 + r + 8 * half;
            const int col = nbase + j * 16 + ln;
            if (FINAL)
                ((float*)Yv)[(size_t)row * N + col] = acc[j][r];
            else
                ((_Float16*)Yv)[(size_t)row * N + col] = (_Float16)acc[j][r];
        }
    }
}

// ---------- Flash-style attention ----------
// Q,K,V: f16 [4096 x 1024] (row = b*2048+s, col = h*64+d)
// mask:  int [B,1,S,S].  Out: f16 [4096 x 1024].
// LDS map: K tile @0 (8K), V^T tile @8192 (8K), P @16384 (8K), mask @24576 (16K)
#define AT_KOFF 0u
#define AT_VOFF 8192u
#define AT_POFF 16384u
#define AT_MOFF 24576u

__global__ void __launch_bounds__(128)
attn_kernel(const _Float16* __restrict__ Q, const _Float16* __restrict__ K,
            const _Float16* __restrict__ V, const int* __restrict__ mask,
            _Float16* __restrict__ O) {
    __shared__ __align__(128) unsigned char SMEM[40960];
    _Float16* Klds  = (_Float16*)(SMEM + AT_KOFF);   // [key][d]
    _Float16* Vtlds = (_Float16*)(SMEM + AT_VOFF);   // [d][key]
    _Float16* Plds  = (_Float16*)(SMEM + AT_POFF);   // [wave][16][64]
    int*      Mlds  = (int*)(SMEM + AT_MOFF);        // [qrow][key]

    const int qt = blockIdx.x & 31;
    const int bh = blockIdx.x >> 5;
    const int h  = bh & 15;
    const int b  = bh >> 4;
    const int qbase = qt * 64;

    const int wave = threadIdx.x >> 5;
    const int lane = threadIdx.x & 31;
    const int half = lane >> 4;
    const int ln   = lane & 15;

    // Q fragments for this wave's 16 rows (A layout row = ln)
    const int qrowA = qbase + wave * 16 + ln;
    const _Float16* qptr = Q + ((size_t)(b * SEQ + qrowA)) * D_MODEL + h * DK;
    const v16h qa0 = load_afrag(qptr + 0, half);    // d 0..31
    const v16h qa1 = load_afrag(qptr + 32, half);   // d 32..63

    float mrow[8], lrow[8];
    v8f acc[4];
#pragma unroll
    for (int r = 0; r < 8; ++r) { mrow[r] = -3.0e38f; lrow[r] = 0.0f; }
#pragma unroll
    for (int jd = 0; jd < 4; ++jd)
#pragma unroll
        for (int r = 0; r < 8; ++r) acc[jd][r] = 0.0f;

    const int t   = threadIdx.x;
    const int key = t >> 1;
    const int dc  = (t & 1) * 32;
    const int* mbase_p = mask + (size_t)b * SEQ * SEQ;

#pragma unroll 1
    for (int kt = 0; kt < SEQ / 64; ++kt) {
        const int kbase = kt * 64;

        // ---- TDM: K tile [64 keys][64 d] and mask tile [64 q][64 keys] ----
        if (threadIdx.x < 32) {
            tdm_load_2d((unsigned long long)(K + ((size_t)(b * SEQ + kbase)) * D_MODEL +
                                             h * DK),
                        AT_KOFF, 1u, 64u, 64u, D_MODEL);
            tdm_load_2d((unsigned long long)(mbase_p + (size_t)qbase * SEQ + kbase),
                        AT_MOFF, 2u, 64u, 64u, SEQ);
        }
        // ---- manual V^T staging (transpose, no TDM path) ----
        const _Float16* vsrc =
            V + ((size_t)(b * SEQ + kbase + key)) * D_MODEL + h * DK + dc;
#pragma unroll
        for (int i = 0; i < 32; ++i)
            Vtlds[(dc + i) * 64 + key] = vsrc[i];
        if (threadIdx.x < 32) __builtin_amdgcn_s_wait_tensorcnt(0);
        __syncthreads();

        // ---- scores: S = Q @ K^T * scale, mask ----
        v8f s[4];
#pragma unroll
        for (int j = 0; j < 4; ++j) {
            v8f c;
#pragma unroll
            for (int r = 0; r < 8; ++r) c[r] = 0.0f;
            const v16h bk0 = *(const v16h*)&Klds[(j * 16 + ln) * 64 + 0 + 16 * half];
            const v16h bk1 = *(const v16h*)&Klds[(j * 16 + ln) * 64 + 32 + 16 * half];
            c = wmma_f16(qa0, bk0, c);
            c = wmma_f16(qa1, bk1, c);
            s[j] = c;
        }
#pragma unroll
        for (int j = 0; j < 4; ++j) {
#pragma unroll
            for (int r = 0; r < 8; ++r) {
                const int qr = wave * 16 + r + 8 * half;        // block-local q row
                float sv = s[j][r] * 0.125f;                    // 1/sqrt(64)
                if (Mlds[qr * 64 + j * 16 + ln] == 0) sv = -3.0e38f;
                s[j][r] = sv;
            }
        }

        // ---- online softmax: row max across the 16-lane half ----
        float tm[8];
#pragma unroll
        for (int r = 0; r < 8; ++r)
            tm[r] = fmaxf(fmaxf(s[0][r], s[1][r]), fmaxf(s[2][r], s[3][r]));
#pragma unroll
        for (int off = 8; off >= 1; off >>= 1)
#pragma unroll
            for (int r = 0; r < 8; ++r)
                tm[r] = fmaxf(tm[r], __shfl_xor(tm[r], off, 32));

        float alpha[8];
#pragma unroll
        for (int r = 0; r < 8; ++r) {
            const float mn = fmaxf(mrow[r], tm[r]);
            alpha[r] = __expf(mrow[r] - mn);
            mrow[r]  = mn;
        }

        float ts[8];
#pragma unroll
        for (int r = 0; r < 8; ++r) ts[r] = 0.0f;
#pragma unroll
        for (int j = 0; j < 4; ++j)
#pragma unroll
            for (int r = 0; r < 8; ++r) {
                const float p = __expf(s[j][r] - mrow[r]);
                s[j][r] = p;
                ts[r] += p;
            }
#pragma unroll
        for (int off = 8; off >= 1; off >>= 1)
#pragma unroll
            for (int r = 0; r < 8; ++r) ts[r] += __shfl_xor(ts[r], off, 32);
#pragma unroll
        for (int r = 0; r < 8; ++r) lrow[r] = lrow[r] * alpha[r] + ts[r];
#pragma unroll
        for (int jd = 0; jd < 4; ++jd)
#pragma unroll
            for (int r = 0; r < 8; ++r) acc[jd][r] *= alpha[r];

        // ---- bounce P through LDS (C-layout -> A-layout), f16 ----
        _Float16* pw = &Plds[wave * 16 * 64];
#pragma unroll
        for (int j = 0; j < 4; ++j)
#pragma unroll
            for (int r = 0; r < 8; ++r)
                pw[(r + 8 * half) * 64 + j * 16 + ln] = (_Float16)s[j][r];

        const _Float16* prow = pw + ln * 64;
        const v16h pa0 = load_afrag(prow + 0, half);    // keys 0..31
        const v16h pa1 = load_afrag(prow + 32, half);   // keys 32..63

        // ---- O += P @ V ----
#pragma unroll
        for (int jd = 0; jd < 4; ++jd) {
            const v16h bv0 = *(const v16h*)&Vtlds[(jd * 16 + ln) * 64 + 0 + 16 * half];
            const v16h bv1 = *(const v16h*)&Vtlds[(jd * 16 + ln) * 64 + 32 + 16 * half];
            acc[jd] = wmma_f16(pa0, bv0, acc[jd]);
            acc[jd] = wmma_f16(pa1, bv1, acc[jd]);
        }
        __syncthreads();
    }

    // ---- normalize and write f16 attn output ----
    float inv[8];
#pragma unroll
    for (int r = 0; r < 8; ++r) inv[r] = 1.0f / lrow[r];
#pragma unroll
    for (int jd = 0; jd < 4; ++jd)
#pragma unroll
        for (int r = 0; r < 8; ++r) {
            const int qr = qbase + wave * 16 + r + 8 * half;
            O[((size_t)(b * SEQ + qr)) * D_MODEL + h * DK + jd * 16 + ln] =
                (_Float16)(acc[jd][r] * inv[r]);
        }
}

// ---------- launch ----------
extern "C" void kernel_launch(void* const* d_in, const int* in_sizes, int n_in,
                              void* d_out, int out_size, void* d_ws, size_t ws_size,
                              hipStream_t stream) {
    (void)in_sizes; (void)n_in; (void)out_size; (void)ws_size;
    const float* q    = (const float*)d_in[0];
    const float* k    = (const float*)d_in[1];
    const float* v    = (const float*)d_in[2];
    const int*   mask = (const int*)d_in[3];
    const float* wq = (const float*)d_in[4];
    const float* bq = (const float*)d_in[5];
    const float* wk = (const float*)d_in[6];
    const float* bk = (const float*)d_in[7];
    const float* wv = (const float*)d_in[8];
    const float* bv = (const float*)d_in[9];
    const float* wo = (const float*)d_in[10];
    const float* bo = (const float*)d_in[11];

    _Float16* Qh = (_Float16*)d_ws;
    _Float16* Kh = Qh + (size_t)MTOT * D_MODEL;
    _Float16* Vh = Kh + (size_t)MTOT * D_MODEL;
    _Float16* Ah = Vh + (size_t)MTOT * D_MODEL;

    const int gemm_blocks = (MTOT / 128) * (D_MODEL / 128);  // 256

    gemm_xwT<0><<<gemm_blocks, 256, 0, stream>>>((const void*)q, wq, bq, (void*)Qh);
    gemm_xwT<0><<<gemm_blocks, 256, 0, stream>>>((const void*)k, wk, bk, (void*)Kh);
    gemm_xwT<0><<<gemm_blocks, 256, 0, stream>>>((const void*)v, wv, bv, (void*)Vh);

    attn_kernel<<<BATCH * NHEAD * (SEQ / 64), 128, 0, stream>>>(Qh, Kh, Vh, mask, Ah);

    gemm_xwT<1><<<gemm_blocks, 256, 0, stream>>>((const void*)Ah, wo, bo, d_out);
}
